// GATModel_807453852438
// MI455X (gfx1250) — compile-verified, matrix operand order checked by reading, not run on hardware
//
#include <hip/hip_runtime.h>
#include <hip/hip_bf16.h>

typedef __attribute__((ext_vector_type(16))) _Float16 v16h;
typedef __attribute__((ext_vector_type(8)))  _Float16 v8h;
typedef __attribute__((ext_vector_type(4)))  _Float16 v4h;
typedef __attribute__((ext_vector_type(8)))  float    v8f;

#define NNODES   20000
#define NEDGES   320000
#define FIN      128
#define HEADS    4
#define HID      64
#define HCDIM    256          // HEADS*HID
#define NGRAPH   64
#define NCLS     4
#define NEG_SLOPE 0.2f
#define ETOT     (NEDGES + NNODES)   // edges + self loops

#define BM 128       // C rows per block = 8 waves x 16
#define GN 4         // 16-col groups per wave -> 16x64 C per wave
#define BSTRIDE 48   // padded smB row stride (halfs): 96B, 16B-aligned frags

// =====================================================================
// WMMA GEMM: C[M,N] = A[M,K] * B[K,N], fp32 in/out, f16 MACs, f32 acc.
// Block = 256 threads (8 waves). Each wave computes a 16x64 C strip
// (4 accumulators; A fragment reused across 4 v_wmma issues).
// Both operands are staged per-block through LDS with coalesced float4
// global loads + one fp32->f16 conversion:
//   smA: [128 rows x 32 K]  f16, 8 KB
//   smB: [64 cols x 32 K]T  f16 (stride 48 halfs), 6 KB
// Fragments are then two contiguous 16B ds_load_b128 runs per operand.
// Per-lane packing follows CDNA5 ISA 7.12.2 (16-bit A 16x32 layout):
//   lane<16  : M=lane,    K offsets {0..7, 16..23}
//   lane>=16 : M=lane-16, K offsets {8..15, 24..31}
// B mirrors with N=lane&15 and the same K pattern.
// =====================================================================
__global__ __launch_bounds__(256) void gemm_wmma_f16(
    const float* __restrict__ A, const float* __restrict__ B,
    float* __restrict__ C, int M, int N, int K)
{
  __shared__ _Float16 smA[BM * 32];             // 8 KB
  __shared__ _Float16 smB[GN * 16 * BSTRIDE];   // 6 KB
  const int tid    = threadIdx.x;
  const int lane   = tid & 31;
  const int wave   = tid >> 5;
  const int mblk   = blockIdx.x * BM;
  const int nbase  = blockIdx.y * (GN * 16);
  const int tmrow  = mblk + wave * 16;          // this wave's first C row
  const bool waveOK = (tmrow < M);              // wave-uniform
  const int l15    = lane & 15;
  const int kshift = (lane >> 4) << 3;          // 0 or 8

  v8f acc[GN] = {};
  for (int k0 = 0; k0 < K; k0 += 32) {
    __syncthreads();                            // prev-iter readers done
    // ---- A slab: [BM x 32] fp32 -> f16 LDS, float4-coalesced ----
#pragma unroll
    for (int i = 0; i < (BM * 32) / (256 * 4); ++i) {
      const int idx4 = i * 256 + tid;
      const int row  = idx4 >> 3;               // 8 float4 per 32-wide row
      const int kq   = (idx4 & 7) << 2;
      const int grow = mblk + row;
      float4 v = make_float4(0.f, 0.f, 0.f, 0.f);
      if (grow < M) v = *(const float4*)(A + (size_t)grow * K + k0 + kq);
      v4h hv; hv[0] = (_Float16)v.x; hv[1] = (_Float16)v.y;
              hv[2] = (_Float16)v.z; hv[3] = (_Float16)v.w;
      *(v4h*)(smA + row * 32 + kq) = hv;
    }
    // ---- B slab: [32 x 64] fp32 -> f16 LDS, stored transposed ----
#pragma unroll
    for (int i = 0; i < (32 * GN * 16) / (256 * 4); ++i) {
      const int idx4 = i * 256 + tid;
      const int kk   = idx4 >> 4;               // 16 float4 per 64-wide row
      const int nq   = (idx4 & 15) << 2;
      const float4 v = *(const float4*)(B + (size_t)(k0 + kk) * N + nbase + nq);
      smB[(nq + 0) * BSTRIDE + kk] = (_Float16)v.x;
      smB[(nq + 1) * BSTRIDE + kk] = (_Float16)v.y;
      smB[(nq + 2) * BSTRIDE + kk] = (_Float16)v.z;
      smB[(nq + 3) * BSTRIDE + kk] = (_Float16)v.w;
    }
    __syncthreads();

    // ---- A fragment: two contiguous 16B runs from LDS ----
    const _Float16* ap = smA + (wave * 16 + l15) * 32 + kshift;
    const v8h alo = *(const v8h*)ap;
    const v8h ahi = *(const v8h*)(ap + 16);
    v16h a;
#pragma unroll
    for (int i = 0; i < 8; ++i) { a[i] = alo[i]; a[i + 8] = ahi[i]; }

    // ---- 4 B fragments from LDS + 4 WMMA issues, A reused ----
#pragma unroll
    for (int g = 0; g < GN; ++g) {
      const _Float16* bp = smB + (g * 16 + l15) * BSTRIDE + kshift;
      const v8h blo = *(const v8h*)bp;
      const v8h bhi = *(const v8h*)(bp + 16);
      v16h b;
#pragma unroll
      for (int i = 0; i < 8; ++i) { b[i] = blo[i]; b[i + 8] = bhi[i]; }
      acc[g] = __builtin_amdgcn_wmma_f32_16x16x32_f16(
          /*neg_a=*/false, a, /*neg_b=*/false, b,
          /*c_mod=*/(short)0, acc[g], /*reuse_a=*/false, /*reuse_b=*/false);
    }
  }

  if (waveOK) {
    // C/D layout: VGPR r -> row r (lanes 0-15) / row r+8 (lanes 16-31)
    const int rbase = tmrow + ((lane >> 4) << 3);
#pragma unroll
    for (int g = 0; g < GN; ++g) {
      const int ncol = nbase + g * 16 + l15;
#pragma unroll
      for (int r = 0; r < 8; ++r)
        C[(size_t)(rbase + r) * N + ncol] = acc[g][r];
    }
  }
}

// ---------------------------------------------------------------------
// alpha_src / alpha_dst per (node, head): dot(h[n,h,:], a_*[h,:])
// ---------------------------------------------------------------------
__global__ void k_alpha(const float* __restrict__ h,
                        const float* __restrict__ a_src,
                        const float* __restrict__ a_dst,
                        float* __restrict__ as, float* __restrict__ ad,
                        int H, int C)
{
  const int i = blockIdx.x * blockDim.x + threadIdx.x;
  if (i >= NNODES * H) return;
  const int n = i / H, hh = i - n * H;
  const float* hp = h + (n * H + hh) * C;
  const float* sp = a_src + hh * C;
  const float* dp = a_dst + hh * C;
  float s = 0.f, d = 0.f;
  for (int c = 0; c < C; ++c) { const float v = hp[c]; s += v * sp[c]; d += v * dp[c]; }
  as[i] = s; ad[i] = d;
}

// ---------------------------------------------------------------------
// CSR build (indexed by dst). Logical edge list = edges then self-loops.
// ---------------------------------------------------------------------
__global__ void k_deg(const int* __restrict__ ei, int* __restrict__ deg)
{
  const int e = blockIdx.x * blockDim.x + threadIdx.x;
  if (e >= ETOT) return;
  const int d = (e < NEDGES) ? ei[NEDGES + e] : (e - NEDGES);
  atomicAdd(&deg[d], 1);
}

__global__ __launch_bounds__(1024) void k_scan_excl(
    const int* __restrict__ deg, int* __restrict__ row_ptr, int n)
{
  __shared__ int sh[1024];
  __shared__ int carry;
  const int tid = threadIdx.x;
  if (tid == 0) { carry = 0; row_ptr[0] = 0; }
  __syncthreads();
  for (int base = 0; base < n; base += 1024) {
    const int i = base + tid;
    sh[tid] = (i < n) ? deg[i] : 0;
    __syncthreads();
    for (int off = 1; off < 1024; off <<= 1) {
      const int t = (tid >= off) ? sh[tid - off] : 0;
      __syncthreads();
      sh[tid] += t;
      __syncthreads();
    }
    const int total = sh[1023];
    const int cbase = carry;
    if (i < n) row_ptr[i + 1] = cbase + sh[tid];
    __syncthreads();
    if (tid == 0) carry = cbase + total;
    __syncthreads();
  }
}

__global__ void k_copy_int(const int* __restrict__ a, int* __restrict__ b, int n)
{
  const int i = blockIdx.x * blockDim.x + threadIdx.x;
  if (i < n) b[i] = a[i];
}

__global__ void k_fill(const int* __restrict__ ei, int* __restrict__ cursor,
                       int* __restrict__ csr_src)
{
  const int e = blockIdx.x * blockDim.x + threadIdx.x;
  if (e >= ETOT) return;
  const int s = (e < NEDGES) ? ei[e]          : (e - NEDGES);
  const int d = (e < NEDGES) ? ei[NEDGES + e] : (e - NEDGES);
  const int pos = atomicAdd(&cursor[d], 1);
  csr_src[pos] = s;
}

// ---------------------------------------------------------------------
// Per-node softmax + aggregation (one block per node, blockDim = H*C):
//   threads 0..H-1 compute per-head max & sum(exp) over incident edges,
//   then all H*C threads gather alpha * h[src] (coalesced over C),
//   fusing +bias and optional ELU. Atomic-free via dst-CSR.
// ---------------------------------------------------------------------
__global__ void k_gat_agg(const float* __restrict__ h,
                          const float* __restrict__ as,
                          const float* __restrict__ ad,
                          const int* __restrict__ row_ptr,
                          const int* __restrict__ csr_src,
                          const float* __restrict__ bias,
                          float* __restrict__ out,
                          int H, int C, int apply_elu)
{
  const int n   = blockIdx.x;
  const int tid = threadIdx.x;
  const int hh  = tid / C;
  const int c   = tid - hh * C;
  __shared__ float m_sh[HEADS], den_sh[HEADS];
  const int beg = row_ptr[n], end = row_ptr[n + 1];

  if (tid < H) {
    const float adn = ad[n * H + tid];
    float mx = -3.0e38f;
    for (int j = beg; j < end; ++j) {
      float v = as[csr_src[j] * H + tid] + adn;
      v = v > 0.f ? v : v * NEG_SLOPE;
      mx = fmaxf(mx, v);
    }
    float den = 0.f;
    for (int j = beg; j < end; ++j) {
      float v = as[csr_src[j] * H + tid] + adn;
      v = v > 0.f ? v : v * NEG_SLOPE;
      den += __expf(v - mx);
    }
    m_sh[tid] = mx; den_sh[tid] = den;
  }
  __syncthreads();

  const float mx   = m_sh[hh];
  const float rden = 1.f / den_sh[hh];
  const float adn  = ad[n * H + hh];
  float acc = 0.f;
  for (int j = beg; j < end; ++j) {
    const int s = csr_src[j];
    float v = as[s * H + hh] + adn;
    v = v > 0.f ? v : v * NEG_SLOPE;
    const float alpha = __expf(v - mx) * rden;
    acc += alpha * h[(s * H + hh) * C + c];
  }
  float o = acc + bias[hh * C + c];
  if (apply_elu) o = o > 0.f ? o : (__expf(o) - 1.f);
  out[(n * H + hh) * C + c] = o;
}

// ---------------------------------------------------------------------
// Global mean pool per graph + 2-layer MLP head, fused. One block/graph.
// Atomic-free & deterministic (batch scan per block, sums tiny).
// ---------------------------------------------------------------------
__global__ __launch_bounds__(64) void k_pool_mlp(
    const float* __restrict__ h3, const int* __restrict__ batch,
    const float* __restrict__ Wc1, const float* __restrict__ bc1,
    const float* __restrict__ Wc2, const float* __restrict__ bc2,
    float* __restrict__ out)
{
  const int g = blockIdx.x;
  const int c = threadIdx.x;            // 0..63
  __shared__ float pooled[HID];
  __shared__ float hid[HID / 2];
  float acc = 0.f; int cnt = 0;
  for (int n = 0; n < NNODES; ++n) {
    if (batch[n] == g) { acc += h3[n * HID + c]; ++cnt; }
  }
  pooled[c] = acc / fmaxf((float)cnt, 1.f);
  __syncthreads();
  if (c < HID / 2) {
    float a = bc1[c];
    for (int k = 0; k < HID; ++k) a += pooled[k] * Wc1[k * (HID / 2) + c];
    hid[c] = fmaxf(a, 0.f);
  }
  __syncthreads();
  if (c < NCLS) {
    float o = bc2[c];
    for (int k = 0; k < HID / 2; ++k) o += hid[k] * Wc2[k * NCLS + c];
    out[g * NCLS + c] = o;
  }
}

// =====================================================================
extern "C" void kernel_launch(void* const* d_in, const int* in_sizes, int n_in,
                              void* d_out, int out_size, void* d_ws, size_t ws_size,
                              hipStream_t stream)
{
  (void)in_sizes; (void)n_in; (void)out_size; (void)ws_size;
  const float* x      = (const float*)d_in[0];
  const int*   ei     = (const int*)  d_in[1];
  const int*   batch  = (const int*)  d_in[2];
  const float* W1     = (const float*)d_in[3];
  const float* a1_src = (const float*)d_in[4];
  const float* a1_dst = (const float*)d_in[5];
  const float* b1     = (const float*)d_in[6];
  const float* W2     = (const float*)d_in[7];
  const float* a2_src = (const float*)d_in[8];
  const float* a2_dst = (const float*)d_in[9];
  const float* b2     = (const float*)d_in[10];
  const float* W3     = (const float*)d_in[11];
  const float* a3_src = (const float*)d_in[12];
  const float* a3_dst = (const float*)d_in[13];
  const float* b3     = (const float*)d_in[14];
  const float* Wc1    = (const float*)d_in[15];
  const float* bc1    = (const float*)d_in[16];
  const float* Wc2    = (const float*)d_in[17];
  const float* bc2    = (const float*)d_in[18];
  float* out = (float*)d_out;

  // ---- carve workspace (256B aligned chunks) ----
  char* p = (char*)d_ws;
  auto carve = [&](size_t bytes) -> void* {
    void* r = (void*)p;
    p += (bytes + 255) & ~(size_t)255;
    return r;
  };
  float* bufA    = (float*)carve((size_t)NNODES * HCDIM * 4);  // gemm out / h pre-agg
  float* bufB    = (float*)carve((size_t)NNODES * HCDIM * 4);  // layer outputs
  float* as      = (float*)carve((size_t)NNODES * HEADS * 4);
  float* ad      = (float*)carve((size_t)NNODES * HEADS * 4);
  int*   deg     = (int*)  carve((size_t)NNODES * 4);
  int*   row_ptr = (int*)  carve((size_t)(NNODES + 1) * 4);
  int*   cursor  = (int*)  carve((size_t)NNODES * 4);
  int*   csr_src = (int*)  carve((size_t)ETOT * 4);

  // ---- CSR build (dst-indexed), shared by all 3 layers ----
  hipMemsetAsync(deg, 0, (size_t)NNODES * 4, stream);
  k_deg      <<<(ETOT + 255) / 256, 256, 0, stream>>>(ei, deg);
  k_scan_excl<<<1, 1024, 0, stream>>>(deg, row_ptr, NNODES);
  k_copy_int <<<(NNODES + 255) / 256, 256, 0, stream>>>(row_ptr, cursor, NNODES);
  k_fill     <<<(ETOT + 255) / 256, 256, 0, stream>>>(ei, cursor, csr_src);

  const dim3 gemmGrid1((NNODES + BM - 1) / BM, HCDIM / (GN * 16));  // (157, 4)
  const dim3 gemmGrid3((NNODES + BM - 1) / BM, HID   / (GN * 16));  // (157, 1)

  // ---- Layer 1: h = x @ W1 ; softmax-agg ; +b1 ; ELU ----
  gemm_wmma_f16<<<gemmGrid1, 256, 0, stream>>>(x, W1, bufA, NNODES, HCDIM, FIN);
  k_alpha<<<(NNODES * HEADS + 255) / 256, 256, 0, stream>>>(
      bufA, a1_src, a1_dst, as, ad, HEADS, HID);
  k_gat_agg<<<NNODES, HEADS * HID, 0, stream>>>(
      bufA, as, ad, row_ptr, csr_src, b1, bufB, HEADS, HID, 1);

  // ---- Layer 2 ----
  gemm_wmma_f16<<<gemmGrid1, 256, 0, stream>>>(bufB, W2, bufA, NNODES, HCDIM, HCDIM);
  k_alpha<<<(NNODES * HEADS + 255) / 256, 256, 0, stream>>>(
      bufA, a2_src, a2_dst, as, ad, HEADS, HID);
  k_gat_agg<<<NNODES, HEADS * HID, 0, stream>>>(
      bufA, as, ad, row_ptr, csr_src, b2, bufB, HEADS, HID, 1);

  // ---- Layer 3 (H=1, no ELU, mean over 1 head == identity) ----
  gemm_wmma_f16<<<gemmGrid3, 256, 0, stream>>>(bufB, W3, bufA, NNODES, HID, HCDIM);
  k_alpha<<<(NNODES * 1 + 255) / 256, 256, 0, stream>>>(
      bufA, a3_src, a3_dst, as, ad, 1, HID);
  k_gat_agg<<<NNODES, HID, 0, stream>>>(
      bufA, as, ad, row_ptr, csr_src, b3, bufB, 1, HID, 0);

  // ---- Pool + classifier head ----
  k_pool_mlp<<<NGRAPH, HID, 0, stream>>>(bufB, batch, Wc1, bc1, Wc2, bc2, out);
}
